// CosFormerAttention_62302795596539
// MI455X (gfx1250) — matrix operand order repaired and checked by reading, not compiled, not run on hardware
//
#include <hip/hip_runtime.h>

typedef __attribute__((ext_vector_type(16))) __bf16 v16bf;
typedef __attribute__((ext_vector_type(8)))  float  v8f;

typedef unsigned int       u32;
typedef unsigned long long u64;
typedef __attribute__((ext_vector_type(4))) u32 tdm_g0_t;
typedef __attribute__((ext_vector_type(8))) int tdm_g1_t;
typedef __attribute__((ext_vector_type(4))) int tdm_g23_t;

#define NHEAD    16
#define HEAD_DIM 64
#define DMODEL   1024
#define BATCH    2
#define TLEN     4096
#define MROWS    (BATCH * TLEN)

// ---------------------------------------------------------------------------
// WMMA helper: D = A(16x32 bf16) * B(32x16 bf16) + C(16x16 f32)
// ---------------------------------------------------------------------------
static __device__ __forceinline__ v8f wmma_bf16(v16bf a, v16bf b, v8f c) {
  return __builtin_amdgcn_wmma_f32_16x16x32_bf16(false, a, false, b, (short)0, c,
                                                 false, false);
}

// A fragment (16x32) from row-major LDS [m][k], row stride rs (elements).
// Contiguous k order used here (A and B alike): lane m = l&15; halves e -> k = k0 + e + 16*(l>=16).
static __device__ __forceinline__ v16bf load_a(const __bf16* p, int row0, int k0, int rs) {
  const int lane = threadIdx.x & 31;
  const __bf16* r = p + (row0 + (lane & 15)) * rs;
  const int kadd = k0 + ((lane >> 4) << 4);
  v16bf f;
#pragma unroll
  for (int e = 0; e < 16; ++e) f[e] = r[kadd + e];
  return f;
}

// B fragment (32x16) from LDS stored transposed Bt[n][k], stride rs.
static __device__ __forceinline__ v16bf load_bT(const __bf16* p, int col0, int k0, int rs) {
  const int lane = threadIdx.x & 31;
  const __bf16* r = p + (col0 + (lane & 15)) * rs + k0 + ((lane >> 4) << 4);
  v16bf f;
#pragma unroll
  for (int e = 0; e < 16; ++e) f[e] = r[e];
  return f;
}

// Same but source is f32 (carried attention state), converted to bf16 on read.
static __device__ __forceinline__ v16bf load_bT_f32(const float* p, int col0, int k0, int rs) {
  const int lane = threadIdx.x & 31;
  const float* r = p + (col0 + (lane & 15)) * rs + k0 + ((lane >> 4) << 4);
  v16bf f;
#pragma unroll
  for (int e = 0; e < 16; ++e) f[e] = (__bf16)r[e];
  return f;
}

// 16x16 f32 accumulator tile <-> LDS stored transposed St[n][m] (stride 64).
static __device__ __forceinline__ v8f load_accT(const float* st, int row0, int col0) {
  const int lane = threadIdx.x & 31;
  const float* r = st + (col0 + (lane & 15)) * 64 + row0 + ((lane >> 4) << 3);
  v8f c;
#pragma unroll
  for (int i = 0; i < 8; ++i) c[i] = r[i];
  return c;
}
static __device__ __forceinline__ void store_accT(float* st, int row0, int col0, v8f c) {
  const int lane = threadIdx.x & 31;
  float* r = st + (col0 + (lane & 15)) * 64 + row0 + ((lane >> 4) << 3);
#pragma unroll
  for (int i = 0; i < 8; ++i) r[i] = c[i];
}

// ---------------------------------------------------------------------------
// Tensor Data Mover: DMA one 64x64 bf16 tile (row stride DMODEL elements in
// global) into LDS. D# per CDNA5 ISA:
//   group0: count=1 | lds_addr | global_addr[56:0] | type=2
//   group1: data_size=1(2B), tensor_dim0/1=64, tile_dim0/1=64, dim0_stride=1024
//   groups 2/3: zero (2-D tile)
// Tracked by TENSORcnt. 6-arg clang-23 builtin form.
// ---------------------------------------------------------------------------
static __device__ __forceinline__ void tdm_load_tile64(const __bf16* gsrc,
                                                       __bf16* ldst) {
  u32 lds = (u32)(u64)ldst;  // generic shared ptr: low 32 bits = LDS offset
  u64 ga = (u64)gsrc;
  tdm_g0_t g0;
  g0[0] = 1u;                                        // count=1, is_restore=0
  g0[1] = lds;                                       // lds_addr
  g0[2] = (u32)ga;                                   // global_addr[31:0]
  g0[3] = (u32)((ga >> 32) & 0x01ffffffu) | (2u << 30);  // addr[56:32], type=2
  tdm_g1_t g1;
  g1[0] = (int)0x00010000u;   // wg_mask=0, data_size=1 (2 bytes)
  g1[1] = (int)(64u << 16);   // tensor_dim0 = 64 (bits 79:48 low half)
  g1[2] = (int)(64u << 16);   // tensor_dim0 hi=0 | tensor_dim1 lo16 = 64
  g1[3] = (int)(64u << 16);   // tensor_dim1 hi=0 | tile_dim0 = 64
  g1[4] = 64;                 // tile_dim1 = 64, tile_dim2 = 0
  g1[5] = DMODEL;             // tensor_dim0_stride = 1024
  g1[6] = 0;                  // dim0_stride hi | dim1_stride lo
  g1[7] = 0;                  // dim1_stride hi
  tdm_g23_t gz = {0, 0, 0, 0};
  tdm_g1_t gz8 = {0, 0, 0, 0, 0, 0, 0, 0};
  __builtin_amdgcn_tensor_load_to_lds(g0, g1, gz, gz, gz8, 0);
}

// ---------------------------------------------------------------------------
// f32 -> bf16 conversion
// ---------------------------------------------------------------------------
__global__ __launch_bounds__(256) void cvt_f32_bf16(const float* __restrict__ in,
                                                    __bf16* __restrict__ out, int n) {
  int i = blockIdx.x * 256 + threadIdx.x;
  if (i < n) out[i] = (__bf16)in[i];
}

// ---------------------------------------------------------------------------
// Tiled bf16 GEMM: C[M,N] = A[M,K] @ W[K,N] + bias, with epilogue modes.
// MODE 0: store bf16 (V projection)
// MODE 1: ReLU then cosFormer cos/sin reweighting -> two bf16 outputs (Q/K)
// MODE 2: store f32 (final output projection)
// ---------------------------------------------------------------------------
template <int MODE>
__global__ __launch_bounds__(128) void gemm_epilogue(
    const __bf16* __restrict__ A, const __bf16* __restrict__ W,
    const float* __restrict__ bias, __bf16* __restrict__ out0,
    __bf16* __restrict__ out1, float* __restrict__ outf, int M, int N, int K) {
  __shared__ __bf16 sA[64 * 32];   // A tile, row-major [m][k]
  __shared__ __bf16 sWt[64 * 32];  // W tile transposed [n][k]
  const int tid = threadIdx.x;
  const int lane = tid & 31, wid = tid >> 5;
  const int rowBase = blockIdx.y * 64;
  const int nBase = blockIdx.x * 64;
  const int row0w = (wid >> 1) * 32, col0w = (wid & 1) * 32;

  const int ar = tid >> 2, akv = (tid & 3) * 8;    // A vector coords (s=0)
  const int wkr = tid >> 3, wnv = (tid & 7) * 8;   // W vector coords (s=0)

  v8f acc[2][2] = {};
  for (int k0 = 0; k0 < K; k0 += 32) {
    __syncthreads();
#pragma unroll
    for (int s = 0; s < 2; ++s) {
      int v = tid + s * 128;
      int r = v >> 2, kv = (v & 3) * 8;
      *(uint4*)&sA[r * 32 + kv] =
          *(const uint4*)&A[(size_t)(rowBase + r) * K + k0 + kv];
      int kr = v >> 3, nv = (v & 7) * 8;
      uint4 wv = *(const uint4*)&W[(size_t)(k0 + kr) * N + nBase + nv];
      const __bf16* h = (const __bf16*)&wv;
#pragma unroll
      for (int e = 0; e < 8; ++e) sWt[(nv + e) * 32 + kr] = h[e];
    }
    // prefetch next K tiles into cache (global_prefetch_b8)
    if (k0 + 32 < K) {
      __builtin_prefetch(&A[(size_t)(rowBase + ar) * K + k0 + 32 + akv], 0, 1);
      __builtin_prefetch(&W[(size_t)(k0 + 32 + wkr) * N + nBase + wnv], 0, 1);
    }
    __syncthreads();
    v16bf a0 = load_a(sA, row0w, 0, 32);
    v16bf a1 = load_a(sA, row0w + 16, 0, 32);
    v16bf b0 = load_bT(sWt, col0w, 0, 32);
    v16bf b1 = load_bT(sWt, col0w + 16, 0, 32);
    acc[0][0] = wmma_bf16(a0, b0, acc[0][0]);
    acc[0][1] = wmma_bf16(a0, b1, acc[0][1]);
    acc[1][0] = wmma_bf16(a1, b0, acc[1][0]);
    acc[1][1] = wmma_bf16(a1, b1, acc[1][1]);
  }

  const int hi8 = ((lane >> 4) << 3), nn = lane & 15;
#pragma unroll
  for (int ti = 0; ti < 2; ++ti)
#pragma unroll
    for (int tj = 0; tj < 2; ++tj) {
      int gn = nBase + col0w + tj * 16 + nn;
      float bv = bias[gn];
#pragma unroll
      for (int r = 0; r < 8; ++r) {
        int gm = rowBase + row0w + ti * 16 + hi8 + r;
        float val = acc[ti][tj][r] + bv;
        if (MODE == 0) {
          out0[(size_t)gm * N + gn] = (__bf16)val;
        } else if (MODE == 1) {
          val = fmaxf(val, 0.0f);  // ReLU
          int t = gm & (TLEN - 1);
          float ang = (1.5707963267948966f / (float)TLEN) * (float)t;
          out0[(size_t)gm * N + gn] = (__bf16)(val * __cosf(ang));
          out1[(size_t)gm * N + gn] = (__bf16)(val * __sinf(ang));
        } else {
          outf[(size_t)gm * N + gn] = val;
        }
      }
    }
}

// ---------------------------------------------------------------------------
// Chunked causal linear-attention scan. One workgroup per (b,h); 64 sequential
// time blocks of 64. Tiles arrive via the Tensor Data Mover; transposed copies
// are built LDS->LDS. All block-level GEMMs via bf16 WMMA; 64x64 f32 states
// S_cos/S_sin live in LDS (transposed) and are updated through the WMMA C path.
// ---------------------------------------------------------------------------
#define ATTN_SMEM_BYTES (8 * 64 * 64 * 2 + 2 * 64 * 64 * 4 + 3 * 64 * 4)

__global__ __launch_bounds__(128) void cosformer_scan(
    const __bf16* __restrict__ gQc, const __bf16* __restrict__ gQs,
    const __bf16* __restrict__ gKc, const __bf16* __restrict__ gKs,
    const __bf16* __restrict__ gV, __bf16* __restrict__ gO) {
  extern __shared__ char smem[];
  __bf16* sQc = (__bf16*)smem;        // [t][d] 64x64
  __bf16* sQs = sQc + 4096;
  __bf16* sKc = sQs + 4096;           // [t][d] (rows are B-operand for scores)
  __bf16* sKs = sKc + 4096;
  __bf16* sKcT = sKs + 4096;          // [d][t] (A-operand for state update)
  __bf16* sKsT = sKcT + 4096;
  __bf16* sVT = sKsT + 4096;          // [d][t] (B-operand, contiguous k)
  __bf16* sSc = sVT + 4096;           // V staging, then masked bf16 scores [i][j]
  float* sStC = (float*)(sSc + 4096); // state S_cos transposed [j][i]
  float* sStS = sStC + 4096;          // state S_sin transposed [j][i]
  float* sZc = sStS + 4096;           // z_cos[64]
  float* sZs = sZc + 64;
  float* sDen = sZs + 64;

  const int tid = threadIdx.x, lane = tid & 31, wid = tid >> 5;
  const int bb = blockIdx.x >> 4, hh = blockIdx.x & 15;
  const int row0w = (wid >> 1) * 32, col0w = (wid & 1) * 32;
  const int hi8 = ((lane >> 4) << 3), nn = lane & 15;

  for (int i = tid; i < 64 * 64; i += 128) { sStC[i] = 0.f; sStS[i] = 0.f; }
  if (tid < 64) { sZc[tid] = 0.f; sZs[tid] = 0.f; }
  __syncthreads();

  for (int ib = 0; ib < TLEN / 64; ++ib) {
    const size_t base = ((size_t)bb * TLEN + (size_t)ib * 64) * DMODEL + hh * HEAD_DIM;

    // ---- TDM: DMA the five 64x64 tiles into LDS (wave 0 issues) --------
    if (wid == 0) {
      tdm_load_tile64(gQc + base, sQc);
      tdm_load_tile64(gQs + base, sQs);
      tdm_load_tile64(gKc + base, sKc);
      tdm_load_tile64(gKs + base, sKs);
      tdm_load_tile64(gV + base, sSc);  // V staged in score buffer
      __builtin_amdgcn_s_wait_tensorcnt(0);
    }
    __syncthreads();

    // ---- build transposed copies LDS->LDS ------------------------------
#pragma unroll
    for (int s = 0; s < 4; ++s) {
      int v = tid + s * 128;
      int row = v >> 3, c8 = (v & 7) * 8;
      uint4 dK = *(const uint4*)(sKc + row * 64 + c8);
      uint4 dS = *(const uint4*)(sKs + row * 64 + c8);
      uint4 dV = *(const uint4*)(sSc + row * 64 + c8);
      const __bf16* hk = (const __bf16*)&dK;
      const __bf16* hs = (const __bf16*)&dS;
      const __bf16* hv = (const __bf16*)&dV;
#pragma unroll
      for (int e = 0; e < 8; ++e) {
        sKcT[(c8 + e) * 64 + row] = hk[e];
        sKsT[(c8 + e) * 64 + row] = hs[e];
        sVT[(c8 + e) * 64 + row] = hv[e];
      }
    }
    __syncthreads();

    // ---- intra-block scores: Qc*Kc^T + Qs*Ks^T -------------------------
    v8f sc[2][2] = {};
#pragma unroll
    for (int kk = 0; kk < 64; kk += 32) {
      v16bf aq0 = load_a(sQc, row0w, kk, 64);
      v16bf aq1 = load_a(sQc, row0w + 16, kk, 64);
      v16bf as0 = load_a(sQs, row0w, kk, 64);
      v16bf as1 = load_a(sQs, row0w + 16, kk, 64);
      v16bf bc0 = load_bT(sKc, col0w, kk, 64);  // Kc[j][d] == [n][k]
      v16bf bc1 = load_bT(sKc, col0w + 16, kk, 64);
      v16bf bs0 = load_bT(sKs, col0w, kk, 64);
      v16bf bs1 = load_bT(sKs, col0w + 16, kk, 64);
      sc[0][0] = wmma_bf16(aq0, bc0, sc[0][0]);
      sc[0][0] = wmma_bf16(as0, bs0, sc[0][0]);
      sc[0][1] = wmma_bf16(aq0, bc1, sc[0][1]);
      sc[0][1] = wmma_bf16(as0, bs1, sc[0][1]);
      sc[1][0] = wmma_bf16(aq1, bc0, sc[1][0]);
      sc[1][0] = wmma_bf16(as1, bs0, sc[1][0]);
      sc[1][1] = wmma_bf16(aq1, bc1, sc[1][1]);
      sc[1][1] = wmma_bf16(as1, bs1, sc[1][1]);
    }
    __syncthreads();  // V staging fully consumed (sVT built) before overwrite
    // causal mask (j <= i) and store bf16 scores
#pragma unroll
    for (int ti = 0; ti < 2; ++ti)
#pragma unroll
      for (int tj = 0; tj < 2; ++tj)
#pragma unroll
        for (int r = 0; r < 8; ++r) {
          int gi = row0w + ti * 16 + hi8 + r;
          int gj = col0w + tj * 16 + nn;
          float v = sc[ti][tj][r];
          if (gj > gi) v = 0.0f;
          sSc[gi * 64 + gj] = (__bf16)v;
        }
    __syncthreads();

    // ---- denominator: masked row-sum + q . z_prev ----------------------
    if (tid < 64) {
      float s = 0.f;
      for (int j = 0; j < 64; ++j) s += (float)sSc[tid * 64 + j];
      for (int d = 0; d < 64; ++d)
        s += (float)sQc[tid * 64 + d] * sZc[d] + (float)sQs[tid * 64 + d] * sZs[d];
      sDen[tid] = fmaxf(s, 1e-6f);
    }

    // ---- numerator: scores@V (intra) + Qc@S_cos + Qs@S_sin (inter) -----
    v8f o[2][2] = {};
#pragma unroll
    for (int kk = 0; kk < 64; kk += 32) {
      v16bf ap0 = load_a(sSc, row0w, kk, 64);
      v16bf ap1 = load_a(sSc, row0w + 16, kk, 64);
      v16bf bv0 = load_bT(sVT, col0w, kk, 64);
      v16bf bv1 = load_bT(sVT, col0w + 16, kk, 64);
      o[0][0] = wmma_bf16(ap0, bv0, o[0][0]);
      o[0][1] = wmma_bf16(ap0, bv1, o[0][1]);
      o[1][0] = wmma_bf16(ap1, bv0, o[1][0]);
      o[1][1] = wmma_bf16(ap1, bv1, o[1][1]);
      v16bf aq0 = load_a(sQc, row0w, kk, 64);
      v16bf aq1 = load_a(sQc, row0w + 16, kk, 64);
      v16bf as0 = load_a(sQs, row0w, kk, 64);
      v16bf as1 = load_a(sQs, row0w + 16, kk, 64);
      v16bf bC0 = load_bT_f32(sStC, col0w, kk, 64);
      v16bf bC1 = load_bT_f32(sStC, col0w + 16, kk, 64);
      v16bf bS0 = load_bT_f32(sStS, col0w, kk, 64);
      v16bf bS1 = load_bT_f32(sStS, col0w + 16, kk, 64);
      o[0][0] = wmma_bf16(aq0, bC0, o[0][0]);
      o[0][0] = wmma_bf16(as0, bS0, o[0][0]);
      o[0][1] = wmma_bf16(aq0, bC1, o[0][1]);
      o[0][1] = wmma_bf16(as0, bS1, o[0][1]);
      o[1][0] = wmma_bf16(aq1, bC0, o[1][0]);
      o[1][0] = wmma_bf16(as1, bS0, o[1][0]);
      o[1][1] = wmma_bf16(aq1, bC1, o[1][1]);
      o[1][1] = wmma_bf16(as1, bS1, o[1][1]);
    }
    __syncthreads();  // S reads done; den ready

    // ---- write normalized output ---------------------------------------
#pragma unroll
    for (int ti = 0; ti < 2; ++ti)
#pragma unroll
      for (int tj = 0; tj < 2; ++tj)
#pragma unroll
        for (int r = 0; r < 8; ++r) {
          int gi = row0w + ti * 16 + hi8 + r;
          int gj = col0w + tj * 16 + nn;
          float val = o[ti][tj][r] / sDen[gi];
          gO[((size_t)bb * TLEN + (size_t)ib * 64 + gi) * DMODEL + hh * HEAD_DIM + gj] =
              (__bf16)val;
        }

    // ---- state update: S += K^T @ V (through WMMA C accumulator) -------
#pragma unroll
    for (int ti = 0; ti < 2; ++ti)
#pragma unroll
      for (int tj = 0; tj < 2; ++tj) {
        int r0 = row0w + ti * 16, c0 = col0w + tj * 16;
        v8f cc = load_accT(sStC, r0, c0);
        v8f cs = load_accT(sStS, r0, c0);
#pragma unroll
        for (int kk = 0; kk < 64; kk += 32) {
          v16bf bv = load_bT(sVT, c0, kk, 64);
          cc = wmma_bf16(load_a(sKcT, r0, kk, 64), bv, cc);
          cs = wmma_bf16(load_a(sKsT, r0, kk, 64), bv, cs);
        }
        store_accT(sStC, r0, c0, cc);
        store_accT(sStS, r0, c0, cs);
      }

    // ---- z update -------------------------------------------------------
    if (tid < 64) {
      float ac = 0.f, as_ = 0.f;
      for (int j = 0; j < 64; ++j) {
        ac += (float)sKcT[tid * 64 + j];
        as_ += (float)sKsT[tid * 64 + j];
      }
      sZc[tid] += ac;
      sZs[tid] += as_;
    }
    __syncthreads();
  }
}

// ---------------------------------------------------------------------------
// Host launcher
// ---------------------------------------------------------------------------
extern "C" void kernel_launch(void* const* d_in, const int* in_sizes, int n_in,
                              void* d_out, int out_size, void* d_ws, size_t ws_size,
                              hipStream_t stream) {
  (void)in_sizes; (void)n_in; (void)out_size; (void)ws_size;
  const float* x  = (const float*)d_in[0];
  const float* Wq = (const float*)d_in[1];
  const float* bq = (const float*)d_in[2];
  const float* Wk = (const float*)d_in[3];
  const float* bk = (const float*)d_in[4];
  const float* Wv = (const float*)d_in[5];
  const float* bv = (const float*)d_in[6];
  const float* Wo = (const float*)d_in[7];
  const float* bo = (const float*)d_in[8];

  char* ws = (char*)d_ws;
  size_t off = 0;
  auto carve = [&](size_t bytes) -> void* {
    void* p = ws + off;
    off += (bytes + 255) & ~(size_t)255;
    return p;
  };
  const size_t actBytes = (size_t)MROWS * DMODEL * 2;  // bf16 activations
  const size_t wBytes   = (size_t)DMODEL * DMODEL * 2; // bf16 weights
  __bf16* xb   = (__bf16*)carve(actBytes);
  __bf16* Wqb  = (__bf16*)carve(wBytes);
  __bf16* Wkb  = (__bf16*)carve(wBytes);
  __bf16* Wvb  = (__bf16*)carve(wBytes);
  __bf16* Wob  = (__bf16*)carve(wBytes);
  __bf16* Qc   = (__bf16*)carve(actBytes);
  __bf16* Qs   = (__bf16*)carve(actBytes);
  __bf16* Kc   = (__bf16*)carve(actBytes);
  __bf16* Ks   = (__bf16*)carve(actBytes);
  __bf16* Vb   = (__bf16*)carve(actBytes);
  __bf16* attn = (__bf16*)carve(actBytes);

  const int nAct = MROWS * DMODEL, nW = DMODEL * DMODEL;
  cvt_f32_bf16<<<(nAct + 255) / 256, 256, 0, stream>>>(x, xb, nAct);
  cvt_f32_bf16<<<(nW + 255) / 256, 256, 0, stream>>>(Wq, Wqb, nW);
  cvt_f32_bf16<<<(nW + 255) / 256, 256, 0, stream>>>(Wk, Wkb, nW);
  cvt_f32_bf16<<<(nW + 255) / 256, 256, 0, stream>>>(Wv, Wvb, nW);
  cvt_f32_bf16<<<(nW + 255) / 256, 256, 0, stream>>>(Wo, Wob, nW);

  dim3 gg(DMODEL / 64, MROWS / 64);
  gemm_epilogue<1><<<gg, 128, 0, stream>>>(xb, Wqb, bq, Qc, Qs, nullptr,
                                           MROWS, DMODEL, DMODEL);
  gemm_epilogue<1><<<gg, 128, 0, stream>>>(xb, Wkb, bk, Kc, Ks, nullptr,
                                           MROWS, DMODEL, DMODEL);
  gemm_epilogue<0><<<gg, 128, 0, stream>>>(xb, Wvb, bv, Vb, nullptr, nullptr,
                                           MROWS, DMODEL, DMODEL);

  cosformer_scan<<<BATCH * NHEAD, 128, ATTN_SMEM_BYTES, stream>>>(Qc, Qs, Kc, Ks,
                                                                  Vb, attn);

  gemm_epilogue<2><<<gg, 128, 0, stream>>>(attn, Wob, bo, nullptr, nullptr,
                                           (float*)d_out, MROWS, DMODEL, DMODEL);
}
